// MultiHeadAttention_49572512530851
// MI455X (gfx1250) — matrix operand order repaired and checked by reading, not compiled, not run on hardware
//
#include <hip/hip_runtime.h>
#include <math.h>

typedef _Float16 v16h __attribute__((ext_vector_type(16)));
typedef _Float16 v8h  __attribute__((ext_vector_type(8)));
typedef float    v8f  __attribute__((ext_vector_type(8)));
typedef float    v4f  __attribute__((ext_vector_type(4)));

#define Bb 8
#define Tt 1024
#define Cc 1024
#define Hh 16
#define HD 64

static __device__ __forceinline__ v16h ld_a16(const _Float16* p0, const _Float16* p1) {
  v8h lo = *(const v8h*)p0;
  v8h hi = *(const v8h*)p1;
  return __builtin_shufflevector(lo, hi, 0,1,2,3,4,5,6,7,8,9,10,11,12,13,14,15);
}

static __device__ __forceinline__ v8f wmma_f16(v16h a, v16h b, v8f c) {
  return __builtin_amdgcn_wmma_f32_16x16x32_f16(false, a, false, b, (short)0, c, false, false);
}

// ---------------- K0: conversions -----------------------------------------
__global__ void k_cvt(const float* __restrict__ src, _Float16* __restrict__ dst, int n) {
  int i = blockIdx.x * blockDim.x + threadIdx.x;
  if (i < n) dst[i] = (_Float16)src[i];
}

// Wq/Wk/Wv [H,C,D] f32 -> [H,D,C] f16
__global__ void k_wtrans(const float* __restrict__ src, _Float16* __restrict__ dst) {
  int i = blockIdx.x * blockDim.x + threadIdx.x;   // H*C*D
  int d = i & (HD - 1);
  int c = (i >> 6) & (Cc - 1);
  int h = i >> 16;
  dst[((size_t)h * HD + d) * Cc + c] = (_Float16)src[i];
}

// ---------------- K1: QKV projections (WMMA GEMM) -------------------------
// blockIdx.x = (((w*8 + b)*16 + h)*64 + ttile), 32 threads (1 wave)
__global__ void k_qkv(const _Float16* __restrict__ xh,
                      const _Float16* __restrict__ wqT,
                      const _Float16* __restrict__ wkT,
                      const _Float16* __restrict__ wvT,
                      _Float16* __restrict__ q,
                      _Float16* __restrict__ k,
                      _Float16* __restrict__ vT) {
  const int lane = threadIdx.x;
  const int lid  = lane & 15;
  const int hi   = lane >> 4;
  int idx = blockIdx.x;
  const int tt = idx & 63;  idx >>= 6;
  const int h  = idx & 15;  idx >>= 4;
  const int b  = idx & 7;   idx >>= 3;
  const int w  = idx;                                   // 0=q 1=k 2=v

  const _Float16* wT = (w == 0) ? wqT : (w == 1) ? wkT : wvT;
  const _Float16* wbase = wT + (size_t)h * HD * Cc;
  const _Float16* xrow  = xh + ((size_t)b * Tt + tt * 16 + lid) * Cc;

  v8f acc[4] = {};
  for (int c0 = 0; c0 < Cc; c0 += 32) {
    v16h a = *(const v16h*)(xrow + c0 + hi * 16);
#pragma unroll
    for (int dt = 0; dt < 4; ++dt) {
      const _Float16* wrow = wbase + (size_t)(dt * 16 + lid) * Cc + c0 + hi * 16;
      v16h bm = *(const v16h*)wrow;
      acc[dt] = wmma_f16(a, bm, acc[dt]);
    }
  }

  const size_t bh = (size_t)b * Hh + h;
  if (w < 2) {                                          // q,k row-major [T,D]
    const float qs = (w == 0) ? 0.125f : 1.0f;          // 1/sqrt(64) folded into q
    _Float16* dst = ((w == 0) ? q : k) + bh * Tt * HD;
#pragma unroll
    for (int dt = 0; dt < 4; ++dt)
#pragma unroll
      for (int i = 0; i < 8; ++i) {
        int trow = tt * 16 + i + hi * 8;
        dst[(size_t)trow * HD + dt * 16 + lid] = (_Float16)(acc[dt][i] * qs);
      }
  } else {                                              // v transposed [D,T]
    _Float16* dst = vT + bh * HD * Tt;
#pragma unroll
    for (int dt = 0; dt < 4; ++dt) {
      v8h pk;
#pragma unroll
      for (int i = 0; i < 8; ++i) pk[i] = (_Float16)acc[dt][i];
      *(v8h*)(dst + (size_t)(dt * 16 + lid) * Tt + tt * 16 + hi * 8) = pk;
    }
  }
}

// ---------------- K2: fused attention (flash-style, writes raw scores) ----
// blockIdx.x = (bh*64 + ttile), 32 threads (1 wave)
__global__ void k_attn(const _Float16* __restrict__ q,
                       const _Float16* __restrict__ k,
                       const _Float16* __restrict__ vT,
                       float* __restrict__ wei,
                       _Float16* __restrict__ ctxh,
                       float* __restrict__ mbuf,
                       float* __restrict__ lbuf) {
  const int lane = threadIdx.x;
  const int lid  = lane & 15;
  const int hi   = lane >> 4;
  int idx = blockIdx.x;
  const int tt = idx & 63; idx >>= 6;
  const int bh = idx;                     // 0..127
  const int b  = bh >> 4;
  const int h  = bh & 15;

  const _Float16* qb = q  + (size_t)bh * Tt * HD;
  const _Float16* kb = k  + (size_t)bh * Tt * HD;
  const _Float16* vb = vT + (size_t)bh * HD * Tt;
  float* weib = wei + (size_t)bh * Tt * Tt;

  const int t0   = tt * 16;
  const int tcol = t0 + lid;

  // B-operand from q (held whole loop): lane = column t, halves = d chunk
  const _Float16* qrow = qb + (size_t)tcol * HD;
  const v16h bq0 = *(const v16h*)(qrow + hi * 16);
  const v16h bq1 = *(const v16h*)(qrow + 32 + hi * 16);

  v8f ctx[4] = {};
  float m = -__builtin_inff();
  float l = 0.f;
  const int ntiles = tt + 1;
  const float NEG = -__builtin_inff();

  for (int st = 0; st < ntiles; st += 2) {
    float P[2][8];
#pragma unroll
    for (int u = 0; u < 2; ++u) {
      const int s_t = st + u;
      if (s_t < ntiles) {                       // wave-uniform branch
        const _Float16* krow = kb + (size_t)(s_t * 16 + lid) * HD;
        v16h a0 = *(const v16h*)(krow + hi * 16);
        v16h a1 = *(const v16h*)(krow + 32 + hi * 16);
        v8f sc = {};
        sc = wmma_f16(a0, bq0, sc);             // S^T tile: row=s, col=t
        sc = wmma_f16(a1, bq1, sc);
#pragma unroll
        for (int i = 0; i < 8; ++i) {
          int srow = s_t * 16 + i + hi * 8;
          P[u][i] = (srow > tcol) ? NEG : sc[i];
        }
        // raw score stream: 512MB, zero reuse before k_norm -> non-temporal
        float* wr = weib + (size_t)tcol * Tt + s_t * 16 + hi * 8;
        v4f r0 = {P[u][0], P[u][1], P[u][2], P[u][3]};
        v4f r1 = {P[u][4], P[u][5], P[u][6], P[u][7]};
        __builtin_nontemporal_store(r0, (v4f*)wr);
        __builtin_nontemporal_store(r1, (v4f*)(wr + 4));
      } else {
#pragma unroll
        for (int i = 0; i < 8; ++i) P[u][i] = NEG;
      }
    }

    // online softmax state (per column t, lane pair keeps identical copies)
    float tm = NEG;
#pragma unroll
    for (int i = 0; i < 8; ++i) tm = fmaxf(tm, fmaxf(P[0][i], P[1][i]));
    tm = fmaxf(tm, __shfl_xor(tm, 16, 32));
    const float mn    = fmaxf(m, tm);
    const float scale = __expf(m - mn);
    float psum = 0.f;
#pragma unroll
    for (int i = 0; i < 8; ++i) {
      P[0][i] = __expf(P[0][i] - mn);
      P[1][i] = __expf(P[1][i] - mn);
      psum += P[0][i] + P[1][i];
    }
    psum += __shfl_xor(psum, 16, 32);
    l = l * scale + psum;
    m = mn;
#pragma unroll
    for (int dt = 0; dt < 4; ++dt)
#pragma unroll
      for (int i = 0; i < 8; ++i) ctx[dt][i] *= scale;

    // Build B-operand P^T [32s x 16t]: half-wave swap lines up C-layout with
    // B-layout K chunks -> only shfl_xor(16) + f16 packs needed.
    v16h bp;
#pragma unroll
    for (int j = 0; j < 8; ++j) {
      float pax = __shfl_xor(P[0][j], 16, 32);
      float pbx = __shfl_xor(P[1][j], 16, 32);
      bp[j]     = (_Float16)((hi == 0) ? P[0][j] : pbx);
      bp[j + 8] = (_Float16)((hi == 0) ? pax     : P[1][j]);
    }

    const int sp = st * 16;
#pragma unroll
    for (int dt = 0; dt < 4; ++dt) {            // ctx^T[d,t] += vT x P^T
      const _Float16* vrow = vb + (size_t)(dt * 16 + lid) * Tt + sp;
      v16h av = *(const v16h*)(vrow + hi * 16);
      ctx[dt] = wmma_f16(av, bp, ctx[dt]);
    }
  }

  const float linv = 1.0f / l;
  if (hi == 0) {
    mbuf[bh * Tt + tcol] = m;
    lbuf[bh * Tt + tcol] = linv;
  }
  // ctx^T C-tile -> row-major ctx f16 [B*T, C]; contiguous 16B per lane.
  // ctxh (16MB) is re-read by k_proj: keep regular temporal (L2-resident).
  _Float16* crow = ctxh + ((size_t)b * Tt + tcol) * Cc + h * HD;
#pragma unroll
  for (int dt = 0; dt < 4; ++dt) {
    v8h pk;
#pragma unroll
    for (int i = 0; i < 8; ++i) pk[i] = (_Float16)(ctx[dt][i] * linv);
    *(v8h*)(crow + dt * 16 + hi * 8) = pk;
  }
}

// ---------------- K3: normalize wei in place (fully non-temporal) ---------
// blockIdx.x = row over B*H*T, 256 threads, float4 each
__global__ void k_norm(float* __restrict__ wei,
                       const float* __restrict__ mbuf,
                       const float* __restrict__ lbuf) {
  const int row = blockIdx.x;
  const int t = row & (Tt - 1);
  const int s = threadIdx.x * 4;
  const float m    = mbuf[row];
  const float linv = lbuf[row];
  float* wr = wei + (size_t)row * Tt + s;
  v4f v = __builtin_nontemporal_load((const v4f*)wr);
  v4f o;
  o.x = (s + 0 <= t) ? __expf(v.x - m) * linv : 0.f;
  o.y = (s + 1 <= t) ? __expf(v.y - m) * linv : 0.f;
  o.z = (s + 2 <= t) ? __expf(v.z - m) * linv : 0.f;
  o.w = (s + 3 <= t) ? __expf(v.w - m) * linv : 0.f;
  __builtin_nontemporal_store(o, (v4f*)wr);
}

// ---------------- K4: output projection (WMMA GEMM + bias) ----------------
// blockIdx.x = (mtile*16 + jtile), 32 threads (1 wave), 16x64 output each
__global__ void k_proj(const _Float16* __restrict__ ctxh,
                       const _Float16* __restrict__ wpH,
                       const float* __restrict__ bias,
                       float* __restrict__ out) {
  const int lane = threadIdx.x;
  const int lid  = lane & 15;
  const int hi   = lane >> 4;
  int idx = blockIdx.x;
  const int jt = idx & 15; idx >>= 4;          // column tile of 64
  const int mt = idx;                          // row tile of 16 over B*T

  const _Float16* arow = ctxh + (size_t)(mt * 16 + lid) * Cc;
  v8f acc[4] = {};
  for (int c0 = 0; c0 < Cc; c0 += 32) {
    v16h a = *(const v16h*)(arow + c0 + hi * 16);
#pragma unroll
    for (int nt = 0; nt < 4; ++nt) {
      // B(K=i, N=j) = Wproj[j][i] -> Wproj rows are already B-layout
      const _Float16* wrow = wpH + (size_t)(jt * 64 + nt * 16 + lid) * Cc + c0 + hi * 16;
      v16h bm = *(const v16h*)wrow;
      acc[nt] = wmma_f16(a, bm, acc[nt]);
    }
  }
#pragma unroll
  for (int nt = 0; nt < 4; ++nt) {
    const int j = jt * 64 + nt * 16 + lid;
    const float bv = bias[j];
#pragma unroll
    for (int i = 0; i < 8; ++i)
      __builtin_nontemporal_store(acc[nt][i] + bv,
                                  out + (size_t)(mt * 16 + i + hi * 8) * Cc + j);
  }
}

// ---------------- host ----------------------------------------------------
extern "C" void kernel_launch(void* const* d_in, const int* in_sizes, int n_in,
                              void* d_out, int out_size, void* d_ws, size_t ws_size,
                              hipStream_t stream) {
  const float* x  = (const float*)d_in[0];
  const float* Wq = (const float*)d_in[1];
  const float* Wk = (const float*)d_in[2];
  const float* Wv = (const float*)d_in[3];
  const float* Wp = (const float*)d_in[4];
  const float* bp = (const float*)d_in[5];

  float* out = (float*)d_out;
  float* wei = out + (size_t)Bb * Tt * Cc;

  char* ws = (char*)d_ws;
  size_t off = 0;
  auto take = [&](size_t bytes) -> char* {
    char* p = ws + off;
    off = (off + bytes + 255) & ~(size_t)255;
    return p;
  };
  _Float16* xh   = (_Float16*)take((size_t)Bb * Tt * Cc * 2);
  _Float16* wqT  = (_Float16*)take((size_t)Hh * HD * Cc * 2);
  _Float16* wkT  = (_Float16*)take((size_t)Hh * HD * Cc * 2);
  _Float16* wvT  = (_Float16*)take((size_t)Hh * HD * Cc * 2);
  _Float16* wpH  = (_Float16*)take((size_t)Cc * Cc * 2);
  _Float16* qf   = (_Float16*)take((size_t)Bb * Hh * Tt * HD * 2);
  _Float16* kf   = (_Float16*)take((size_t)Bb * Hh * Tt * HD * 2);
  _Float16* vTf  = (_Float16*)take((size_t)Bb * Hh * HD * Tt * 2);
  _Float16* ctxh = (_Float16*)take((size_t)Bb * Tt * Cc * 2);
  float*    mbuf = (float*)take((size_t)Bb * Hh * Tt * 4);
  float*    lbuf = (float*)take((size_t)Bb * Hh * Tt * 4);

  { int n = Bb * Tt * Cc;  k_cvt<<<n / 256, 256, 0, stream>>>(x, xh, n); }
  { int n = Hh * Cc * HD;
    k_wtrans<<<n / 256, 256, 0, stream>>>(Wq, wqT);
    k_wtrans<<<n / 256, 256, 0, stream>>>(Wk, wkT);
    k_wtrans<<<n / 256, 256, 0, stream>>>(Wv, wvT); }
  { int n = Cc * Cc;       k_cvt<<<n / 256, 256, 0, stream>>>(Wp, wpH, n); }

  k_qkv <<<3 * Bb * Hh * 64, 32, 0, stream>>>(xh, wqT, wkT, wvT, qf, kf, vTf);
  k_attn<<<Bb * Hh * 64,     32, 0, stream>>>(qf, kf, vTf, wei, ctxh, mbuf, lbuf);
  k_norm<<<Bb * Hh * Tt,    256, 0, stream>>>(wei, mbuf, lbuf);
  k_proj<<<(Bb * Tt / 16) * (Cc / 64), 32, 0, stream>>>(ctxh, wpH, bp, out);
}
